// Attention_7662221656252
// MI455X (gfx1250) — matrix operand order repaired and checked by reading, not compiled, not run on hardware
//
#include <hip/hip_runtime.h>

typedef unsigned short u16;
typedef __attribute__((ext_vector_type(16))) __bf16 v16bf;
typedef __attribute__((ext_vector_type(16))) unsigned short v16u;
typedef __attribute__((ext_vector_type(8))) float v8f;

#define NN 4096
#define LSTR 40
#define PI_F 3.14159265358979323846f

// ---------- bf16 helpers (RNE) ----------
__device__ __forceinline__ u16 f2bf(float f) {
  unsigned u = __builtin_bit_cast(unsigned, f);
  unsigned r = u + 0x7FFFu + ((u >> 16) & 1u);
  return (u16)(r >> 16);
}
__device__ __forceinline__ float bf2f(u16 h) {
  return __builtin_bit_cast(float, ((unsigned)h) << 16);
}
__device__ __forceinline__ float gelu_f(float v) {
  return 0.5f * v * (1.0f + erff(v * 0.70710678118654752440f));
}

// ---------- WMMA fragment loads from LDS (contiguous k = 16h + j, same order for A and B) ----------
// A (16x32, MxK): lane L holds row M=(L&15); half j -> k=(L>=16?16:0)+j.
__device__ __forceinline__ v16bf load_fragA(const u16* s, int row, int lane) {
  const u16* p = s + row * LSTR;
  int kA = (lane & 16) ? 16 : 0;
  v16u t;
#pragma unroll
  for (int j = 0; j < 16; ++j) t[j] = p[kA + j];
  return __builtin_bit_cast(v16bf, t);
}
// B (32x16, KxN): lane L holds col N=(L&15); half j -> k=(L>=16?16:0)+j.
// sX is stored [n][k] (k contiguous).
__device__ __forceinline__ v16bf load_fragB(const u16* s, int col, int lane) {
  const u16* p = s + col * LSTR + ((lane & 16) ? 16 : 0);
  v16u t;
#pragma unroll
  for (int j = 0; j < 16; ++j) t[j] = p[j];
  return __builtin_bit_cast(v16bf, t);
}

// ---------- generic conv1x1 GEMM: Y = act(W @ X + bias) + resid ----------
// W: bf16 MxK row-major. X: per-batch KxNN (bf16 or fp32, compile-time).
// Double-buffered LDS, one barrier per K-step.
template <bool XF32>
__global__ __launch_bounds__(256) void gemm_wmma(
    const u16* __restrict__ W, const void* __restrict__ X,
    float* __restrict__ Yf, u16* __restrict__ Yb,
    const float* __restrict__ bias, const float* __restrict__ resid,
    int M, int K, long xStride, long yStride, long rStride, int doGelu) {
  __shared__ u16 sW[2][128 * LSTR];
  __shared__ u16 sX[2][64 * LSTR];
  const int bz = blockIdx.z;
  const int n0 = blockIdx.x * 64;
  const int m0 = blockIdx.y * 128;
  const int tid = threadIdx.x;
  const int wave = tid >> 5, lane = tid & 31;
  const int wm = (wave & 3) * 32;
  const int wn = (wave >> 2) * 32;
  const u16* Xb16 = (const u16*)X;
  const float* Xf32 = (const float*)X;

  // tile loaders
  auto load_tiles = [&](int k0, int s) {
    // W tile: 128 rows x 32 k; one b128 global load + one b128 LDS store
    {
      int row = tid >> 1, kh = (tid & 1) * 16;
      const uint4* g = (const uint4*)(W + (long)(m0 + row) * K + k0 + kh);
      uint4 v = *g;
      *(uint4*)(&sW[s][row * LSTR + kh]) = v;
    }
    // X tile: 32 k x 64 n, transposed into [n][k].
    // thread: k = tid>>3, n8 = (tid&7)*8 ; contiguous-n vector global load.
    {
      int k = tid >> 3, n8 = (tid & 7) * 8;
      long gbase = (long)bz * xStride + (long)(k0 + k) * NN + n0 + n8;
      u16 h[8];
      if constexpr (XF32) {
        const float4* p = (const float4*)(Xf32 + gbase);
        float4 a = p[0], b = p[1];
        h[0] = f2bf(a.x); h[1] = f2bf(a.y); h[2] = f2bf(a.z); h[3] = f2bf(a.w);
        h[4] = f2bf(b.x); h[5] = f2bf(b.y); h[6] = f2bf(b.z); h[7] = f2bf(b.w);
      } else {
        union { uint4 u; u16 hh[8]; } uu;
        uu.u = *(const uint4*)(Xb16 + gbase);
#pragma unroll
        for (int i = 0; i < 8; ++i) h[i] = uu.hh[i];
      }
#pragma unroll
      for (int i = 0; i < 8; ++i) sX[s][(n8 + i) * LSTR + k] = h[i];
    }
  };

  v8f acc[2][2];
#pragma unroll
  for (int i = 0; i < 2; ++i)
#pragma unroll
    for (int j = 0; j < 2; ++j)
#pragma unroll
      for (int r = 0; r < 8; ++r) acc[i][j][r] = 0.0f;

  const int nk = K >> 5;
  load_tiles(0, 0);
  for (int it = 0; it < nk; ++it) {
    __syncthreads();
    if (it + 1 < nk) load_tiles((it + 1) << 5, (it + 1) & 1);
    const int s = it & 1;
    const u16* cw = sW[s];
    const u16* cx = sX[s];
#pragma unroll
    for (int i = 0; i < 2; ++i) {
      v16bf a = load_fragA(cw, wm + i * 16, lane);
#pragma unroll
      for (int j = 0; j < 2; ++j) {
        v16bf b = load_fragB(cx, wn + j * 16, lane);
        acc[i][j] = __builtin_amdgcn_wmma_f32_16x16x32_bf16(
            false, a, false, b, (short)0, acc[i][j], false, false);
      }
    }
  }
  // store: C layout lane L, vgpr r -> M = r + (L>=16?8:0), N = L&15
#pragma unroll
  for (int i = 0; i < 2; ++i) {
#pragma unroll
    for (int j = 0; j < 2; ++j) {
      int mB = m0 + wm + i * 16 + ((lane & 16) ? 8 : 0);
      int nB = n0 + wn + j * 16 + (lane & 15);
#pragma unroll
      for (int r = 0; r < 8; ++r) {
        int m = mB + r;
        float v = acc[i][j][r];
        if (bias) v += bias[m];
        if (doGelu) v = gelu_f(v);
        if (resid) v += resid[(long)bz * rStride + (long)m * NN + nB];
        long yi = (long)bz * yStride + (long)m * NN + nB;
        if (Yf) Yf[yi] = v;
        if (Yb) Yb[yi] = f2bf(v);
      }
    }
  }
}

// ---------- partial conv (first 64 channels, SAME, KSxKS), bf16 output ----------
template <int KS>
__global__ __launch_bounds__(256) void pconv_kernel(
    const float* __restrict__ in, const float* __restrict__ w,
    u16* __restrict__ out) {
  constexpr int P = KS / 2;
  constexpr int D = 16 + 2 * P;
  __shared__ float tile[16 * D * D];
  const int b = blockIdx.y;
  const int t = blockIdx.x;
  const int y0 = (t >> 2) * 16, x0 = (t & 3) * 16;
  const int tid = threadIdx.x;
  const int py = tid >> 4, px = tid & 15;

  float acc[64];
#pragma unroll
  for (int o = 0; o < 64; ++o) acc[o] = 0.0f;

  for (int icb = 0; icb < 4; ++icb) {
    for (int idx = tid; idx < 16 * D * D; idx += 256) {
      int ic = idx / (D * D);
      int rem = idx - ic * (D * D);
      int yy = rem / D, xx = rem - yy * D;
      int gy = y0 + yy - P, gx = x0 + xx - P;
      float v = 0.0f;
      if (gy >= 0 && gy < 64 && gx >= 0 && gx < 64)
        v = in[((long)b * 256 + icb * 16 + ic) * NN + gy * 64 + gx];
      tile[idx] = v;
    }
    __syncthreads();
    for (int ic = 0; ic < 16; ++ic) {
      for (int ky = 0; ky < KS; ++ky) {
        for (int kx = 0; kx < KS; ++kx) {
          float tv = tile[ic * D * D + (py + ky) * D + (px + kx)];
          const float* wp = w + (long)(icb * 16 + ic) * KS * KS + ky * KS + kx;
#pragma unroll
          for (int o = 0; o < 64; ++o)
            acc[o] = fmaf(wp[(long)o * 64 * KS * KS], tv, acc[o]);
        }
      }
    }
    __syncthreads();
  }
#pragma unroll
  for (int o = 0; o < 64; ++o)
    out[((long)b * 256 + o) * NN + (y0 + py) * 64 + (x0 + px)] = f2bf(acc[o]);
}

// ---------- copy/convert channels 64..255 (pconv pass-through) ----------
__global__ void cvt_copy_upper(const float* __restrict__ in, u16* __restrict__ o1,
                               u16* __restrict__ o2) {
  long idx = (long)blockIdx.x * blockDim.x + threadIdx.x;
  const long total = 16L * 192 * NN;
  if (idx >= total) return;
  long b = idx / (192L * NN);
  long r = idx - b * 192L * NN;
  long ch = 64 + r / NN;
  long n = r & (NN - 1);
  long off = (b * 256 + ch) * NN + n;
  u16 v = f2bf(in[off]);
  o1[off] = v;
  if (o2) o2[off] = v;
}

// ---------- fp32 -> bf16 elementwise ----------
__global__ void cvt_bf16(const float* __restrict__ in, u16* __restrict__ out, int n) {
  int idx = blockIdx.x * blockDim.x + threadIdx.x;
  if (idx < n) out[idx] = f2bf(in[idx]);
}

// ---------- attention per (b,h): Gram-collapsed dual branch ----------
__global__ __launch_bounds__(256) void attn_kernel(
    const u16* __restrict__ qkv, u16* __restrict__ pin,
    const float* __restrict__ temp1, const float* __restrict__ temp2,
    const float* __restrict__ tw1, const float* __restrict__ tw2) {
  const int h = blockIdx.x, b = blockIdx.y;
  const int tid = threadIdx.x;
  const int c = tid >> 4, d = tid & 15;
  const u16* Q = qkv + ((long)b * 768 + h * 32 + 16) * NN;
  const u16* Kp = Q + 256L * NN;
  const u16* V = Q + 512L * NN;

  __shared__ float Gqk[16][16], Gqq[16][16], Gkk[16][16];
  __shared__ float A1[16][16], Ms[16][16];
  __shared__ float aRe[9][9], aIm[9][9];
  __shared__ float qn_s[16], kn_s[16], fqn[9], fkn[9];

  // Phase 1: Gram matrices over n (each thread owns one (c,d) entry)
  {
    const unsigned* pqc = (const unsigned*)(Q + (long)c * NN);
    const unsigned* pkd = (const unsigned*)(Kp + (long)d * NN);
    const unsigned* pqd = (const unsigned*)(Q + (long)d * NN);
    const unsigned* pkc = (const unsigned*)(Kp + (long)c * NN);
    float s1 = 0.f, s2 = 0.f, s3 = 0.f;
    for (int n2 = 0; n2 < NN / 2; ++n2) {
      unsigned a = pqc[n2], e = pkd[n2], f = pqd[n2], g = pkc[n2];
      float qc0 = bf2f((u16)a), qc1 = bf2f((u16)(a >> 16));
      float kd0 = bf2f((u16)e), kd1 = bf2f((u16)(e >> 16));
      float qd0 = bf2f((u16)f), qd1 = bf2f((u16)(f >> 16));
      float kc0 = bf2f((u16)g), kc1 = bf2f((u16)(g >> 16));
      s1 += qc0 * kd0 + qc1 * kd1;
      s2 += qc0 * qd0 + qc1 * qd1;
      s3 += kc0 * kd0 + kc1 * kd1;
    }
    Gqk[c][d] = s1;
    Gqq[c][d] = s2;
    Gkk[c][d] = s3;
  }
  __syncthreads();

  const float t1 = temp1[h], t2 = temp2[h];

  if (tid < 16) {
    qn_s[tid] = fmaxf(sqrtf(fmaxf(Gqq[tid][tid], 0.f)), 1e-12f);
    kn_s[tid] = fmaxf(sqrtf(fmaxf(Gkk[tid][tid], 0.f)), 1e-12f);
  }
  __syncthreads();

  A1[c][d] = Gqk[c][d] / (qn_s[c] * kn_s[d]) * t1;
  __syncthreads();

  // spatial softmax (rows) * tw2
  if (tid < 16) {
    float mx = -1e30f;
    for (int j = 0; j < 16; ++j) mx = fmaxf(mx, A1[tid][j]);
    float s = 0.f;
    for (int j = 0; j < 16; ++j) {
      float e = expf(A1[tid][j] - mx);
      A1[tid][j] = e;
      s += e;
    }
    float inv = 1.f / s;
    for (int j = 0; j < 16; ++j)
      A1[tid][j] = A1[tid][j] * inv * tw2[h * 256 + tid * 16 + j];
  }

  // FFT-domain norms: ||qf_c||^2 = sum_{j,l} cos(pi*c*(j-l)/8) * Gqq[j][l]
  if (tid < 9) {
    float sq = 0.f, sk = 0.f;
    for (int j = 0; j < 16; ++j)
      for (int l = 0; l < 16; ++l) {
        float wv = cosf(PI_F * 0.125f * (float)(tid * (j - l)));
        sq += wv * Gqq[j][l];
        sk += wv * Gkk[j][l];
      }
    fqn[tid] = fmaxf(sqrtf(fmaxf(sq, 0.f)), 1e-12f);
    fkn[tid] = fmaxf(sqrtf(fmaxf(sk, 0.f)), 1e-12f);
  }
  __syncthreads();

  // gf = F * Gqk * F^T (9x9 complex), normalized & scaled
  if (tid < 81) {
    int cc = tid / 9, dd = tid - cc * 9;
    float re = 0.f, im = 0.f;
    for (int j = 0; j < 16; ++j)
      for (int l = 0; l < 16; ++l) {
        float ang = PI_F * 0.125f * (float)(cc * j + dd * l);
        float g = Gqk[j][l];
        re += cosf(ang) * g;
        im -= sinf(ang) * g;
      }
    float sc = t2 / (fqn[cc] * fkn[dd]);
    aRe[cc][dd] = re * sc;
    aIm[cc][dd] = im * sc;
  }
  __syncthreads();

  // separate softmax over real & imag rows, then * tw1
  if (tid < 9) {
    float mr = -1e30f, mi = -1e30f;
    for (int j = 0; j < 9; ++j) {
      mr = fmaxf(mr, aRe[tid][j]);
      mi = fmaxf(mi, aIm[tid][j]);
    }
    float sr = 0.f, si = 0.f;
    for (int j = 0; j < 9; ++j) {
      float er = expf(aRe[tid][j] - mr), ei = expf(aIm[tid][j] - mi);
      aRe[tid][j] = er;
      aIm[tid][j] = ei;
      sr += er;
      si += ei;
    }
    float ir = 1.f / sr, ii = 1.f / si;
    for (int j = 0; j < 9; ++j) {
      float tw = tw1[h * 81 + tid * 9 + j];
      aRe[tid][j] = aRe[tid][j] * ir * tw;
      aIm[tid][j] = aIm[tid][j] * ii * tw;
    }
  }
  __syncthreads();

  // M[j][l] = (1/16)(ReT0 + (-1)^j ReT8 + 2*sum_{c=1..7}(ReTc cos - ImTc sin)),
  // with T[c][l] = sum_d a[c][d] * e^{-i pi d l / 8}
  {
    const int jj = c, ll = d;
    float m = 0.f;
    for (int cc = 0; cc < 9; ++cc) {
      float tre = 0.f, tim = 0.f;
      for (int dd = 0; dd < 9; ++dd) {
        float bang = PI_F * 0.125f * (float)(dd * ll);
        float cb = cosf(bang), sb = sinf(bang);
        tre += aRe[cc][dd] * cb + aIm[cc][dd] * sb;
        tim += aIm[cc][dd] * cb - aRe[cc][dd] * sb;
      }
      if (cc == 0) {
        m += tre;
      } else if (cc == 8) {
        m += (jj & 1) ? -tre : tre;
      } else {
        float th = PI_F * 0.125f * (float)(cc * jj);
        m += 2.f * (tre * cosf(th) - tim * sinf(th));
      }
    }
    Ms[jj][ll] = m * (1.f / 16.f);
  }
  __syncthreads();

  // Phase 3: out1 = A1 @ v ; lx = Ms @ v ; write into proj input (B,256,NN)
  for (int n = tid; n < NN; n += 256) {
    float vv[16];
#pragma unroll
    for (int l = 0; l < 16; ++l) vv[l] = bf2f(V[(long)l * NN + n]);
#pragma unroll
    for (int row = 0; row < 16; ++row) {
      float s1 = 0.f, s2 = 0.f;
#pragma unroll
      for (int l = 0; l < 16; ++l) {
        s1 += A1[row][l] * vv[l];
        s2 += Ms[row][l] * vv[l];
      }
      pin[((long)b * 256 + 16 * h + row) * NN + n] = f2bf(s2);        // lx
      pin[((long)b * 256 + 128 + 16 * h + row) * NN + n] = f2bf(s1);  // out1
    }
  }
}

extern "C" void kernel_launch(void* const* d_in, const int* in_sizes, int n_in,
                              void* d_out, int out_size, void* d_ws, size_t ws_size,
                              hipStream_t stream) {
  const float* x = (const float*)d_in[0];
  const float* pc3a_w = (const float*)d_in[1];
  const float* hm_conv1_w = (const float*)d_in[2];
  const float* hm_proj2_w = (const float*)d_in[3];
  const float* hm_proj2_b = (const float*)d_in[4];
  const float* pc5_w = (const float*)d_in[5];
  const float* hm_conv2_w = (const float*)d_in[6];
  const float* fuse_w = (const float*)d_in[7];
  const float* qkv_pc3_w = (const float*)d_in[8];
  const float* qkv_w = (const float*)d_in[9];
  const float* proj_w = (const float*)d_in[10];
  const float* proj_b = (const float*)d_in[11];
  const float* temp1 = (const float*)d_in[12];
  const float* temp2 = (const float*)d_in[13];
  const float* tw1 = (const float*)d_in[14];
  const float* tw2 = (const float*)d_in[15];

  char* ws = (char*)d_ws;
  const size_t MB = 1ull << 20;
  u16* W_hm1 = (u16*)ws;              // 65536
  u16* W_hm2 = W_hm1 + 65536;         // 65536
  u16* W_hmc2 = W_hm1 + 131072;       // 65536
  u16* W_fuse = W_hm1 + 196608;       // 196608
  u16* W_qkv = W_hm1 + 393216;        // 196608
  u16* W_proj = W_hm1 + 589824;       // 65536
  u16* XA = (u16*)(ws + 2 * MB);      // 32MB bf16 (B,256,NN)
  u16* XB = (u16*)(ws + 34 * MB);     // 32MB
  u16* CPR = (u16*)(ws + 66 * MB);    // 96MB bf16 (B,768,NN)
  float* HX = (float*)(ws + 162 * MB);// 64MB fp32 (B,256,NN)
  u16* HQ = XA;                       // reuse (XA dead)
  u16* QKV = (u16*)(ws + 34 * MB);    // 96MB bf16 (B,768,NN), reuse XB+CPR
  u16* PIN = (u16*)(ws + 130 * MB);   // 32MB bf16 (B,256,NN)

  // 1. weights -> bf16
  cvt_bf16<<<(65536 + 255) / 256, 256, 0, stream>>>(hm_conv1_w, W_hm1, 65536);
  cvt_bf16<<<(65536 + 255) / 256, 256, 0, stream>>>(hm_proj2_w, W_hm2, 65536);
  cvt_bf16<<<(65536 + 255) / 256, 256, 0, stream>>>(hm_conv2_w, W_hmc2, 65536);
  cvt_bf16<<<(196608 + 255) / 256, 256, 0, stream>>>(fuse_w, W_fuse, 196608);
  cvt_bf16<<<(196608 + 255) / 256, 256, 0, stream>>>(qkv_w, W_qkv, 196608);
  cvt_bf16<<<(65536 + 255) / 256, 256, 0, stream>>>(proj_w, W_proj, 65536);

  // 2-4. partial convs + pass-through channels
  pconv_kernel<3><<<dim3(16, 16), 256, 0, stream>>>(x, pc3a_w, XA);
  pconv_kernel<5><<<dim3(16, 16), 256, 0, stream>>>(x, pc5_w, XB);
  {
    long total = 16L * 192 * NN;
    cvt_copy_upper<<<(unsigned)((total + 255) / 256), 256, 0, stream>>>(x, XA, XB);
  }

  // 5-7. cx/px/rx -> CPR (gelu)
  gemm_wmma<false><<<dim3(64, 2, 16), 256, 0, stream>>>(
      W_hm1, XA, nullptr, CPR, nullptr, nullptr,
      256, 256, 256L * NN, 768L * NN, 0, 1);
  gemm_wmma<true><<<dim3(64, 2, 16), 256, 0, stream>>>(
      W_hm2, x, nullptr, CPR + 256L * NN, hm_proj2_b, nullptr,
      256, 256, 256L * NN, 768L * NN, 0, 1);
  gemm_wmma<false><<<dim3(64, 2, 16), 256, 0, stream>>>(
      W_hmc2, XB, nullptr, CPR + 512L * NN, nullptr, nullptr,
      256, 256, 256L * NN, 768L * NN, 0, 1);

  // 8. hx = fuse @ CPR + x (fp32)
  gemm_wmma<false><<<dim3(64, 2, 16), 256, 0, stream>>>(
      W_fuse, CPR, HX, nullptr, nullptr, x,
      256, 768, 768L * NN, 256L * NN, 256L * NN, 0);

  // 9-10. qkv partial conv on hx
  pconv_kernel<3><<<dim3(16, 16), 256, 0, stream>>>(HX, qkv_pc3_w, HQ);
  {
    long total = 16L * 192 * NN;
    cvt_copy_upper<<<(unsigned)((total + 255) / 256), 256, 0, stream>>>(HX, HQ, nullptr);
  }

  // 11. qkv GEMM
  gemm_wmma<false><<<dim3(64, 6, 16), 256, 0, stream>>>(
      W_qkv, HQ, nullptr, QKV, nullptr, nullptr,
      768, 256, 256L * NN, 768L * NN, 0, 0);

  // 12. attention -> PIN
  attn_kernel<<<dim3(8, 16), 256, 0, stream>>>(QKV, PIN, temp1, temp2, tw1, tw2);

  // 13. proj -> d_out (fp32, bias)
  gemm_wmma<false><<<dim3(64, 2, 16), 256, 0, stream>>>(
      W_proj, PIN, (float*)d_out, nullptr, proj_b, nullptr,
      256, 256, 256L * NN, 256L * NN, 0, 0);
}
